// MultiHeadAttention_18803366822467
// MI455X (gfx1250) — matrix operand order repaired and checked by reading, not compiled, not run on hardware
//
#include <hip/hip_runtime.h>
#include <hip/hip_bf16.h>

// ---------------------------------------------------------------------------
// MHA forward for MI455X (gfx1250, wave32, WMMA + async-to-LDS).
// Shapes: B=2, S=2048, D=1024, H=16, DK=64.
// Outputs (concatenated in d_out, fp32): output [B,S,D], attn_weights [B,H,S,S].
// ---------------------------------------------------------------------------

typedef __attribute__((ext_vector_type(16))) __bf16 v16bf;
typedef __attribute__((ext_vector_type(8)))  __bf16 v8bf;
typedef __attribute__((ext_vector_type(8)))  float  v8f;
typedef __attribute__((ext_vector_type(4)))  float  v4f;
typedef int v4i_gcc __attribute__((vector_size(16)));  // matches builtin proto

constexpr int Bc  = 2;
constexpr int Sc  = 2048;
constexpr int Dc  = 1024;
constexpr int Hc  = 16;
constexpr int DKc = 64;

#if defined(__AMDGCN__) && __has_builtin(__builtin_amdgcn_global_load_async_to_lds_b128)
#define HAVE_ASYNC_LDS 1
#else
#define HAVE_ASYNC_LDS 0
#endif

// Copy 16 bytes (8 bf16) global -> LDS. Async (ASYNCcnt) when available.
__device__ inline void copy16_g2l(const __bf16* g, __bf16* l) {
#if HAVE_ASYNC_LDS
  __builtin_amdgcn_global_load_async_to_lds_b128(
      (v4i_gcc*)g, (v4i_gcc*)l, /*offset=*/0, /*cpol=*/0);
#else
  *(v8bf*)l = *(const v8bf*)g;
#endif
}

__device__ inline void async_wait_all() {
#if HAVE_ASYNC_LDS
#if __has_builtin(__builtin_amdgcn_s_wait_asynccnt)
  __builtin_amdgcn_s_wait_asynccnt(0);
#else
  asm volatile("s_wait_asynccnt 0" ::: "memory");
#endif
#endif
}

__device__ inline v8f wmma_bf16(v16bf a, v16bf b, v8f c) {
  // D = A(16x32 bf16) * B(32x16 bf16) + C(16x16 f32)
  return __builtin_amdgcn_wmma_f32_16x16x32_bf16(
      /*neg_a=*/false, a, /*neg_b=*/false, b,
      /*c_mod=*/(short)0, c, /*reuse_a=*/false, /*reuse_b=*/false);
}

// ---------------------------------------------------------------------------
// fp32 -> bf16 elementwise convert (vectorized, 8 elems/thread).
// ---------------------------------------------------------------------------
__global__ __launch_bounds__(256)
void cvt_f32_bf16_kernel(const float* __restrict__ in, __bf16* __restrict__ out, int n) {
  const int i = (blockIdx.x * 256 + threadIdx.x) * 8;
  if (i + 8 <= n) {
    const v4f a = *(const v4f*)(in + i);
    const v4f b = *(const v4f*)(in + i + 4);
    v8bf o;
#pragma unroll
    for (int j = 0; j < 4; ++j) { o[j] = (__bf16)a[j]; o[4 + j] = (__bf16)b[j]; }
    *(v8bf*)(out + i) = o;
  }
}

// ---------------------------------------------------------------------------
// GEMM:  C[m,n] = sum_k A[m,k] * W[n,k] + bias[n]   (A, W both bf16)
//   M = B*S = 4096, N = K = D = 1024
// OMODE: 0 -> head-split bf16 out [B,H,S,DK]
//        1 -> flat fp32 out [M,N]
//        2 -> head-split TRANSPOSED bf16 out [B,H,DK,S]  (for V^T)
// Tile 128x128, K-step 64, 8 waves, 16 WMMAs per K-step.
// Double-buffered LDS (2 x 32 KB) with async global->LDS copies:
// issue tile t+1, compute tile t, s_wait_asynccnt + one barrier per step.
// ---------------------------------------------------------------------------
template <int OMODE>
__global__ __launch_bounds__(256)
void gemm_wmma_kernel(const __bf16* __restrict__ A, const __bf16* __restrict__ Wb,
                      const float* __restrict__ bias, void* __restrict__ Out) {
  constexpr int N = Dc;
  constexpr int K = Dc;
  constexpr int BM = 128, BN = 128, BK = 64;

  __shared__ __align__(128) __bf16 As[2][BM][BK];   // 2 x 16 KB
  __shared__ __align__(128) __bf16 Bs[2][BN][BK];   // 2 x 16 KB

  const int tid  = threadIdx.x;
  const int lane = tid & 31;
  const int wv   = tid >> 5;
  const int m0   = blockIdx.y * BM;
  const int n0   = blockIdx.x * BN;
  const int wm   = (wv & 3) * 32;   // wave row offset in tile
  const int wn   = (wv >> 2) * 64;  // wave col offset in tile
  const int lm   = lane & 15;
  const int klo  = (lane < 16) ? 0 : 8;    // A-frag K sub-offset
  const int kb   = (lane < 16) ? 0 : 16;   // B-frag K half
  const int roff = (lane < 16) ? 0 : 8;    // C/D row half

  // Per-thread fill slice: row fr, 32-wide K chunk fc (64 B = 4 x b128 copies).
  const int fr = tid >> 1;
  const int fc = (tid & 1) * 32;
  const __bf16* gA = A  + (size_t)(m0 + fr) * K + fc;
  const __bf16* gW = Wb + (size_t)(n0 + fr) * K + fc;

  auto issue_tile = [&](int buf, int k0) {
    const __bf16* a = gA + k0;
    const __bf16* w = gW + k0;
    __bf16* la = &As[buf][fr][fc];
    __bf16* lb = &Bs[buf][fr][fc];
#pragma unroll
    for (int j = 0; j < 32; j += 8) {
      copy16_g2l(a + j, la + j);
      copy16_g2l(w + j, lb + j);
    }
  };

  issue_tile(0, 0);
  async_wait_all();
  __syncthreads();

  v8f acc[2][4] = {};

  for (int k0 = 0; k0 < K; k0 += BK) {
    const int cur = (k0 >> 6) & 1;
    if (k0 + BK < K) {
      issue_tile(cur ^ 1, k0 + BK);                     // overlap with compute
      if (k0 + 2 * BK < K) __builtin_prefetch(gW + k0 + 2 * BK, 0, 1);
    }

    // ---- two K=32 sub-steps on the current LDS tile ----
#pragma unroll
    for (int kc = 0; kc < BK; kc += 32) {
      // A-frag (16x32): lane -> row lm; elem e -> k = kc + kb + e (contiguous)
      v16bf af[2];
#pragma unroll
      for (int tm = 0; tm < 2; ++tm) {
        const __bf16* p = &As[cur][wm + tm * 16 + lm][kc];
        v8bf lo = *(const v8bf*)(p + kb);
        v8bf hi = *(const v8bf*)(p + 8 + kb);
        v16bf a;
#pragma unroll
        for (int i = 0; i < 8; ++i) { a[i] = lo[i]; a[8 + i] = hi[i]; }
        af[tm] = a;
      }
      // B-frag (32x16): lane -> col lm; elem e -> k = kc + kb + e (contiguous)
      v16bf bfv[4];
#pragma unroll
      for (int tn = 0; tn < 4; ++tn)
        bfv[tn] = *(const v16bf*)&Bs[cur][wn + tn * 16 + lm][kc + kb];

#pragma unroll
      for (int tm = 0; tm < 2; ++tm)
#pragma unroll
        for (int tn = 0; tn < 4; ++tn)
          acc[tm][tn] = wmma_bf16(af[tm], bfv[tn], acc[tm][tn]);
    }

    async_wait_all();   // next tile's copies have landed (this wave)
    __syncthreads();    // all waves: done reading cur, done filling next
  }

  // ---- epilogue: C/D layout: elem r -> row roff+r, lane -> col lm ----
#pragma unroll
  for (int tm = 0; tm < 2; ++tm) {
#pragma unroll
    for (int tn = 0; tn < 4; ++tn) {
#pragma unroll
      for (int r = 0; r < 8; ++r) {
        const int m = m0 + wm + tm * 16 + roff + r;
        const int n = n0 + wn + tn * 16 + lm;
        const float val = acc[tm][tn][r] + bias[n];
        if constexpr (OMODE == 0) {
          const int b  = m >> 11;            // m / S
          const int s  = m & (Sc - 1);
          const int h  = n >> 6;             // n / DK
          const int dk = n & (DKc - 1);
          ((__bf16*)Out)[(((size_t)(b * Hc + h)) * Sc + s) * DKc + dk] = (__bf16)val;
        } else if constexpr (OMODE == 1) {
          ((float*)Out)[(size_t)m * N + n] = val;
        } else {  // OMODE == 2: V^T head-split [B,H,DK,S]
          const int b  = m >> 11;
          const int s  = m & (Sc - 1);
          const int h  = n >> 6;
          const int dk = n & (DKc - 1);
          ((__bf16*)Out)[(((size_t)(b * Hc + h)) * DKc + dk) * Sc + s] = (__bf16)val;
        }
      }
    }
  }
}

// ---------------------------------------------------------------------------
// Attention core: one workgroup per (b, h, 16-row q chunk).
//   scores(16x2048) = q(16x64) @ k^T * 0.125   -> LDS (fp32, 128 KB)
//   softmax rows in LDS (+mask), write attn_weights once
//   O(16x64) = P(16x2048) @ v(2048x64)         -> workspace bf16 [B,S,D]
// V is provided TRANSPOSED ([B,H,DK,S]) so B-frags are contiguous v16bf loads.
// Cross-wave O reduction is a fixed-order serial sum (deterministic, no atomics).
// Dynamic LDS: 128 KB scores + 1 KB reduce + 32 KB O-partials = 164864 B
// (within CDNA5's 320 KB/WGP budget).
// ---------------------------------------------------------------------------
__global__ __launch_bounds__(256)
void attn_kernel(const __bf16* __restrict__ q, const __bf16* __restrict__ k,
                 const __bf16* __restrict__ vt, const int* __restrict__ mask,
                 float* __restrict__ attnw, __bf16* __restrict__ oout) {
  extern __shared__ __align__(128) char smem[];
  float (*sc)[Sc]         = (float(*)[Sc])smem;                          // 16 x 2048
  float (*red)[16]        = (float(*)[16])(smem + 16 * Sc * 4);          // 16 x 16
  float (*opart)[16][DKc] = (float(*)[16][DKc])(smem + 16 * Sc * 4 + 1024);  // 8 x 16 x 64

  const int tid  = threadIdx.x;
  const int lane = tid & 31;
  const int wv   = tid >> 5;
  const int lm   = lane & 15;
  const int klo  = (lane < 16) ? 0 : 8;
  const int kb   = (lane < 16) ? 0 : 16;
  const int roff = (lane < 16) ? 0 : 8;

  const int q0 = blockIdx.x * 16;
  const int h  = blockIdx.y;
  const int b  = blockIdx.z;
  const size_t headoff = ((size_t)(b * Hc + h)) * Sc * DKc;
  const __bf16* qh  = q + headoff;
  const __bf16* kh  = k + headoff;
  const __bf16* vth = vt + headoff;  // [DK][S] layout within head

  // ---- Q fragments (shared by all score tiles): two K=32 chunks of DK=64 ----
  v16bf qf[2];
#pragma unroll
  for (int c = 0; c < 2; ++c) {
    const __bf16* p = qh + (size_t)(q0 + lm) * DKc + c * 32;
    v8bf lo = *(const v8bf*)(p + kb);
    v8bf hi = *(const v8bf*)(p + 8 + kb);
    v16bf a;
#pragma unroll
    for (int i = 0; i < 8; ++i) { a[i] = lo[i]; a[8 + i] = hi[i]; }
    qf[c] = a;
  }

  // ---- scores: each wave sweeps every 8th 16-col tile ----
  const float scale = 0.125f;  // 1/sqrt(64)
  for (int nt = wv; nt < Sc / 16; nt += 8) {
    const int n0 = nt * 16;
    // B-frag = k^T: B[d][n] = k[n][d]; per lane n = n0+lm, d = (chunk) + kb + e
    const __bf16* kp = kh + (size_t)(n0 + lm) * DKc;
    v16bf b0 = *(const v16bf*)(kp + kb);        // d in [kb, kb+16)
    v16bf b1 = *(const v16bf*)(kp + 32 + kb);   // d in [32+kb, 32+kb+16)
    v8f accs = {};
    accs = wmma_bf16(qf[0], b0, accs);
    accs = wmma_bf16(qf[1], b1, accs);
#pragma unroll
    for (int r = 0; r < 8; ++r)
      sc[roff + r][n0 + lm] = accs[r] * scale;
  }
  __syncthreads();

  // ---- softmax: 16 threads per row ----
  const int row = tid >> 4;
  const int sub = tid & 15;
  const int* mrow = mask + b * Sc;
  float mx = -3.4e38f;
  for (int n = sub; n < Sc; n += 16) {
    float s = sc[row][n];
    if (mrow[n] == 0) { s = -__builtin_inff(); sc[row][n] = s; }
    mx = fmaxf(mx, s);
  }
  red[row][sub] = mx;
  __syncthreads();
  if (sub == 0) {
    float m2 = red[row][0];
#pragma unroll
    for (int i = 1; i < 16; ++i) m2 = fmaxf(m2, red[row][i]);
    red[row][0] = m2;
  }
  __syncthreads();
  mx = red[row][0];
  float sum = 0.f;
  for (int n = sub; n < Sc; n += 16) {
    const float e = __expf(sc[row][n] - mx);
    sc[row][n] = e;
    sum += e;
  }
  __syncthreads();  // everyone done reading red[row][0]
  red[row][sub] = sum;
  __syncthreads();
  if (sub == 0) {
    float s2 = 0.f;
#pragma unroll
    for (int i = 0; i < 16; ++i) s2 += red[row][i];
    red[row][0] = s2;
  }
  __syncthreads();
  const float inv = 1.f / red[row][0];
  float* wout = attnw + ((size_t)(b * Hc + h) * Sc + q0 + row) * Sc;
  for (int n = sub; n < Sc; n += 16) {
    const float p = sc[row][n] * inv;
    sc[row][n] = p;    // keep normalized P for the PV GEMM
    wout[n] = p;       // required output: attn_weights
  }
  __syncthreads();

  // ---- O = P @ V : each wave owns a 256-deep K stripe (8 x K=32 steps) ----
  typedef __attribute__((ext_vector_type(8))) float v8ff;  // 32B LDS vector load
  v8f oacc[4] = {};
  for (int kc = wv * 256; kc < wv * 256 + 256; kc += 32) {
    // A-frag from P (LDS fp32 -> bf16): row lm; two contiguous 8-float runs
    const v8ff r0 = *(const v8ff*)&sc[lm][kc + kb];
    const v8ff r1 = *(const v8ff*)&sc[lm][kc + 8 + kb];
    v16bf pa;
#pragma unroll
    for (int i = 0; i < 8; ++i) { pa[i] = (__bf16)r0[i]; pa[8 + i] = (__bf16)r1[i]; }
#pragma unroll
    for (int tn = 0; tn < 4; ++tn) {
      // B-frag from V^T: lane -> col d = tn*16+lm; k(seq) = kc + kb + e contiguous
      const v16bf vb = *(const v16bf*)&vth[(size_t)(tn * 16 + lm) * Sc + kc + kb];
      oacc[tn] = wmma_bf16(pa, vb, oacc[tn]);
    }
  }
  // each wave writes its partial O tile to its own LDS slab (conflict-free)
#pragma unroll
  for (int tn = 0; tn < 4; ++tn)
#pragma unroll
    for (int r = 0; r < 8; ++r)
      opart[wv][roff + r][tn * 16 + lm] = oacc[tn][r];
  __syncthreads();

  // deterministic fixed-order reduction over the 8 wave slabs, then store bf16
  for (int i = tid; i < 16 * DKc; i += 256) {
    const int rr = i >> 6;
    const int d  = i & 63;
    float s = 0.f;
#pragma unroll
    for (int w = 0; w < 8; ++w) s += opart[w][rr][d];
    oout[((size_t)(b * Sc) + q0 + rr) * Dc + h * DKc + d] = (__bf16)s;
  }
}

// ---------------------------------------------------------------------------
// Launch: bf16 pre-convert (Q,K,V,W*) -> 3 QKV GEMMs -> attention -> O GEMM.
// Workspace layout (bf16): Qb,Kb,Vb (8 MB ea) | Wq,Wk,Wv,Wo (2 MB ea)
//                          | qws,kws,vtws (8 MB ea); ows aliases Qb (dead).
// ---------------------------------------------------------------------------
extern "C" void kernel_launch(void* const* d_in, const int* in_sizes, int n_in,
                              void* d_out, int out_size, void* d_ws, size_t ws_size,
                              hipStream_t stream) {
  (void)in_sizes; (void)n_in; (void)out_size; (void)ws_size;

  const float* Q    = (const float*)d_in[0];
  const float* K    = (const float*)d_in[1];
  const float* V    = (const float*)d_in[2];
  const int*   mask = (const int*)d_in[3];
  const float* Wq   = (const float*)d_in[4];
  const float* bq   = (const float*)d_in[5];
  const float* Wk   = (const float*)d_in[6];
  const float* bk   = (const float*)d_in[7];
  const float* Wv   = (const float*)d_in[8];
  const float* bv   = (const float*)d_in[9];
  const float* Wo   = (const float*)d_in[10];
  const float* bo   = (const float*)d_in[11];

  const size_t actElems = (size_t)Bc * Sc * Dc;  // 4,194,304
  const size_t wElems   = (size_t)Dc * Dc;       // 1,048,576

  __bf16* Qb   = (__bf16*)d_ws;
  __bf16* Kb   = Qb + actElems;
  __bf16* Vb   = Kb + actElems;
  __bf16* Wqb  = Vb + actElems;
  __bf16* Wkb  = Wqb + wElems;
  __bf16* Wvb  = Wkb + wElems;
  __bf16* Wob  = Wvb + wElems;
  __bf16* qws  = Wob + wElems;
  __bf16* kws  = qws + actElems;
  __bf16* vtws = kws + actElems;   // V^T [B,H,DK,S]
  __bf16* ows  = Qb;               // reuse: Qb is dead after the Q projection

  float* out   = (float*)d_out;
  float* attnw = out + actElems;   // attn_weights region

  // ---- bf16 pre-conversion ----
  const int actBlocks = (int)(actElems / 8 / 256);  // 2048
  const int wBlocks   = (int)(wElems / 8 / 256);    // 512
  cvt_f32_bf16_kernel<<<actBlocks, 256, 0, stream>>>(Q,  Qb,  (int)actElems);
  cvt_f32_bf16_kernel<<<actBlocks, 256, 0, stream>>>(K,  Kb,  (int)actElems);
  cvt_f32_bf16_kernel<<<actBlocks, 256, 0, stream>>>(V,  Vb,  (int)actElems);
  cvt_f32_bf16_kernel<<<wBlocks,   256, 0, stream>>>(Wq, Wqb, (int)wElems);
  cvt_f32_bf16_kernel<<<wBlocks,   256, 0, stream>>>(Wk, Wkb, (int)wElems);
  cvt_f32_bf16_kernel<<<wBlocks,   256, 0, stream>>>(Wv, Wvb, (int)wElems);
  cvt_f32_bf16_kernel<<<wBlocks,   256, 0, stream>>>(Wo, Wob, (int)wElems);

  // ---- projections ----
  const dim3 gemmGrid(Dc / 128, (Bc * Sc) / 128);  // (8, 32)
  gemm_wmma_kernel<0><<<gemmGrid, 256, 0, stream>>>(Qb, Wqb, bq, qws);
  gemm_wmma_kernel<0><<<gemmGrid, 256, 0, stream>>>(Kb, Wkb, bk, kws);
  gemm_wmma_kernel<2><<<gemmGrid, 256, 0, stream>>>(Vb, Wvb, bv, vtws);

  // ---- attention ----
  const dim3 attnGrid(Sc / 16, Hc, Bc);  // (128, 16, 2)
  const size_t smemBytes = (size_t)16 * Sc * 4 + 1024 + (size_t)8 * 16 * DKc * 4;  // 164864
  attn_kernel<<<attnGrid, 256, smemBytes, stream>>>(qws, kws, vtws, mask, attnw, ows);

  // ---- output projection ----
  gemm_wmma_kernel<1><<<gemmGrid, 256, 0, stream>>>(ows, Wob, bo, out);
}
